// LlamaAttention_82652350644924
// MI455X (gfx1250) — matrix operand order repaired and checked by reading, not compiled, not run on hardware
//
#include <hip/hip_runtime.h>
#include <hip/hip_bf16.h>

// ---------------------------------------------------------------------------
// Problem constants (match the reference)
// ---------------------------------------------------------------------------
#define BB   2
#define QQ   1024
#define HH   32
#define KVH_ 8
#define DD   64
#define HID_ 2048
#define LCK_ 3
static constexpr float SCALE = 0.125f;   // 1/sqrt(64)

typedef __bf16 v16bf __attribute__((ext_vector_type(16)));
typedef __bf16 v8bf  __attribute__((ext_vector_type(8)));
typedef float  v8f   __attribute__((ext_vector_type(8)));

// ---------------------------------------------------------------------------
// Helpers
// ---------------------------------------------------------------------------
__device__ __forceinline__ v8f zero_v8f() {
  v8f z;
#pragma unroll
  for (int i = 0; i < 8; ++i) z[i] = 0.0f;
  return z;
}

__device__ __forceinline__ v8f wmma_bf16(v16bf a, v16bf b, v8f c) {
  // D = A(16x32 bf16) * B(32x16 bf16) + C(16x16 f32)
  return __builtin_amdgcn_wmma_f32_16x16x32_bf16(
      /*neg_a=*/false, a, /*neg_b=*/false, b,
      /*c_mod=*/(short)0, c, /*reuse_a=*/false, /*reuse_b=*/false);
}

// A fragment: 16(M) x 32(K) bf16 tile from row-major [M x ld] starting at k0.
// ISA layout: lanes 0-15 -> M=lane, K in {k0..k0+7, k0+16..k0+23};
//             lanes 16-31 -> M=lane-16, K shifted by +8.
__device__ __forceinline__ v16bf load_a_frag(const __bf16* base, int ld,
                                             int k0, int lane) {
  const int m  = lane & 15;
  const int kh = (lane >> 4) << 3;  // 0 or 8
  const __bf16* p = base + (size_t)m * ld + k0 + kh;
  const v8bf lo = *reinterpret_cast<const v8bf*>(p);
  const v8bf hi = *reinterpret_cast<const v8bf*>(p + 16);
  v16bf a;
#pragma unroll
  for (int i = 0; i < 8; ++i) { a[i] = lo[i]; a[i + 8] = hi[i]; }
  return a;
}

// B fragment: 32(K) x 16(N) where B[k][n] = src[n*ld + k]  (i.e. B = srcT).
// ISA layout: lanes 0-15 -> N=lane, K in {k0..k0+7, k0+16..k0+23}; lanes 16-31 -> K shifted by +8.
__device__ __forceinline__ v16bf load_bT_frag(const __bf16* src, int ld,
                                              int k0, int lane) {
  const int n  = lane & 15;
  const int kk = k0 + ((lane >> 4) << 3);
  const __bf16* p = src + (size_t)n * ld + kk;
  const v8bf lo = *reinterpret_cast<const v8bf*>(p);
  const v8bf hi = *reinterpret_cast<const v8bf*>(p + 16);
  v16bf b;
#pragma unroll
  for (int i = 0; i < 8; ++i) { b[i] = lo[i]; b[i + 8] = hi[i]; }
  return b;
}

// Reductions over the 16-lane group (wave32: xor 1,2,4,8 stays in-half).
__device__ __forceinline__ float redmax16(float v) {
  v = fmaxf(v, __shfl_xor(v, 1, 32));
  v = fmaxf(v, __shfl_xor(v, 2, 32));
  v = fmaxf(v, __shfl_xor(v, 4, 32));
  v = fmaxf(v, __shfl_xor(v, 8, 32));
  return v;
}
__device__ __forceinline__ float redsum16(float v) {
  v += __shfl_xor(v, 1, 32);
  v += __shfl_xor(v, 2, 32);
  v += __shfl_xor(v, 4, 32);
  v += __shfl_xor(v, 8, 32);
  return v;
}

// ---------------------------------------------------------------------------
// 1) fp32 -> bf16 conversion
// ---------------------------------------------------------------------------
__global__ void f32_to_bf16_kernel(const float* __restrict__ in,
                                   __bf16* __restrict__ out, int n) {
  for (int i = blockIdx.x * blockDim.x + threadIdx.x; i < n;
       i += gridDim.x * blockDim.x)
    out[i] = (__bf16)in[i];
}

// ---------------------------------------------------------------------------
// 2) GEMM: C[M,N] (fp32) = A[M,K](bf16) @ W[N,K](bf16)^T
//    One wave computes a 32x64 C tile (2 A-frags x 4 B-frags, 8 WMMA/k-step).
//    Wave->tile mapping keeps tm fast-varying so the 8 waves of a block share
//    the same 64-column weight tile (B fetched once into WGP$, hit 7 more
//    times) -> ~8x less L2 traffic on the big operand.
// ---------------------------------------------------------------------------
__global__ __launch_bounds__(256) void gemm_bf16_nt(
    const __bf16* __restrict__ A, const __bf16* __restrict__ W,
    float* __restrict__ C, int M, int N, int K) {
  const int gw   = (blockIdx.x * blockDim.x + threadIdx.x) >> 5;
  const int lane = threadIdx.x & 31;
  const int mt   = M >> 5;                 // M tiles of 32 (fast index)
  const int nt   = N >> 6;                 // N tiles of 64 (slow index)
  const int tm   = gw % mt, tn = gw / mt;
  if (tn >= nt) return;
  const __bf16* A0 = A + (size_t)tm * 32 * K;
  const __bf16* A1 = A0 + (size_t)16 * K;
  const __bf16* W0 = W + (size_t)tn * 64 * K;
  v8f acc[2][4];
#pragma unroll
  for (int mi = 0; mi < 2; ++mi)
#pragma unroll
    for (int t = 0; t < 4; ++t) acc[mi][t] = zero_v8f();

  for (int k0 = 0; k0 < K; k0 += 32) {
    const v16bf a0 = load_a_frag(A0, K, k0, lane);
    const v16bf a1 = load_a_frag(A1, K, k0, lane);
#pragma unroll
    for (int t = 0; t < 4; ++t) {
      const v16bf b = load_bT_frag(W0 + (size_t)(t * 16) * K, K, k0, lane);
      acc[0][t] = wmma_bf16(a0, b, acc[0][t]);
      acc[1][t] = wmma_bf16(a1, b, acc[1][t]);
    }
    __builtin_prefetch(A0 + k0 + 64, 0, 3);  // near-cache streaming prefetch
    __builtin_prefetch(A1 + k0 + 64, 0, 3);
  }
  const int rb = (lane >> 4) << 3, cn = lane & 15;
#pragma unroll
  for (int mi = 0; mi < 2; ++mi) {
#pragma unroll
    for (int i = 0; i < 8; ++i) {
      float* Crow =
          C + (size_t)(tm * 32 + mi * 16 + rb + i) * N + tn * 64 + cn;
      Crow[0]  = acc[mi][0][i];
      Crow[16] = acc[mi][1][i];
      Crow[32] = acc[mi][2][i];
      Crow[48] = acc[mi][3][i];
    }
  }
}

// ---------------------------------------------------------------------------
// 3) RoPE (positions = position_ids + LCK) + scatter [B,Q,NH*D] -> [B,NH,Q,D]
// ---------------------------------------------------------------------------
__global__ void rope_scatter_kernel(const float* __restrict__ x,
                                    const int* __restrict__ pos_ids,
                                    __bf16* __restrict__ dst, int NH,
                                    int do_rope) {
  const int total = BB * QQ * NH * DD;
  for (int idx = blockIdx.x * blockDim.x + threadIdx.x; idx < total;
       idx += gridDim.x * blockDim.x) {
    const int d = idx & 63;
    const int h = (idx >> 6) % NH;
    const int q = (idx / (64 * NH)) % QQ;
    const int b = idx / (64 * NH * QQ);
    const float v = x[idx];
    float outv = v;
    if (do_rope) {
      const int dh = d & 31;
      const float inv_freq = __powf(10000.0f, -(float)(2 * dh) * (1.0f / 64.0f));
      const float ang = (float)(pos_ids[b * QQ + q] + LCK_) * inv_freq;
      float s, c;
      __sincosf(ang, &s, &c);
      const float vr = x[(d < 32) ? (idx + 32) : (idx - 32)];
      outv = (d < 32) ? (v * c - vr * s) : (v * c + vr * s);
    }
    dst[(((size_t)b * NH + h) * QQ + q) * DD + d] = (__bf16)outv;
  }
}

// ---------------------------------------------------------------------------
// 4) Diagonal logits: dlog[b,h,q,n] = scale * q_rope . diag_k[n],
//    diag_k = {cache_k[1], cache_k[2], k_new[kv-head h/4]}
// ---------------------------------------------------------------------------
__global__ void diag_logits_kernel(const __bf16* __restrict__ qb,
                                   const float* __restrict__ cache_k,
                                   const __bf16* __restrict__ kb,
                                   float* __restrict__ dlog) {
  const int idx = blockIdx.x * blockDim.x + threadIdx.x;  // (b,h,q) flat
  if (idx >= BB * HH * QQ) return;
  const int qg = idx % QQ;
  const int h  = (idx / QQ) % HH;
  const int b  = idx / (QQ * HH);
  const size_t slice = (size_t)BB * HH * QQ * DD;
  const __bf16* qp  = qb + (size_t)idx * DD;
  const float*  ck1 = cache_k + slice + (size_t)idx * DD;
  const float*  ck2 = cache_k + 2 * slice + (size_t)idx * DD;
  const __bf16* kp  = kb + (((size_t)b * KVH_ + (h >> 2)) * QQ + qg) * DD;
  float s0 = 0.f, s1 = 0.f, s2 = 0.f;
#pragma unroll 8
  for (int d = 0; d < DD; ++d) {
    const float qv = (float)qp[d];
    s0 += qv * ck1[d];
    s1 += qv * ck2[d];
    s2 += qv * (float)kp[d];
  }
  dlog[(size_t)idx * 3 + 0] = s0 * SCALE;
  dlog[(size_t)idx * 3 + 1] = s1 * SCALE;
  dlog[(size_t)idx * 3 + 2] = s2 * SCALE;
}

// ---------------------------------------------------------------------------
// 5) Flash attention: one wave = 16-query tile of one (b,h).
//    Streams 32-key chunks of K0/V0, online softmax in fp32, then folds the
//    3 diagonal logits + diag-V vectors, writes attn [B,Q,H*D] bf16.
//    NOTE: no block barriers in the loop (waves have different trip counts);
//    per-wave LDS ordering is guaranteed by the ISA (DS ops in-order).
// ---------------------------------------------------------------------------
__global__ __launch_bounds__(256) void flash_attn_kernel(
    const __bf16* __restrict__ qb,    // [B,H,Q,D] rope'd
    const __bf16* __restrict__ k0b,   // [B,H,Q,D]
    const __bf16* __restrict__ v0b,   // [B,H,Q,D]
    const float*  __restrict__ dlog,  // [B,H,Q,3]
    const float*  __restrict__ cache_v,  // [LCK,B,H,Q,D] fp32
    const __bf16* __restrict__ vnew,  // [B,KVH,Q,D]
    __bf16* __restrict__ attn_out) {  // [B,Q,H*D]
  __shared__ alignas(16) __bf16 lds_vt[8][DD * 32];  // per-wave V^T  [dim][key]
  __shared__ alignas(16) __bf16 lds_p[8][16 * 32];   // per-wave P    [row][key]
  const int wid  = threadIdx.x >> 5;
  const int lane = threadIdx.x & 31;
  const int NQT  = QQ / 16;
  const int gw   = blockIdx.x * 8 + wid;
  if (gw >= BB * HH * NQT) return;
  const int qt = gw % NQT;
  const int h  = (gw / NQT) % HH;
  const int b  = gw / (NQT * HH);
  const int q0 = qt * 16;
  const size_t headoff = ((size_t)b * HH + h) * QQ * DD;

  const __bf16* qbase = qb + headoff + (size_t)q0 * DD;
  const v16bf aq0 = load_a_frag(qbase, DD, 0, lane);
  const v16bf aq1 = load_a_frag(qbase, DD, 32, lane);
  const __bf16* kbase = k0b + headoff;
  const __bf16* vbase = v0b + headoff;

  v8f o0 = zero_v8f(), o1 = zero_v8f(), o2 = zero_v8f(), o3 = zero_v8f();
  float m_i[8], l_i[8];
#pragma unroll
  for (int i = 0; i < 8; ++i) { m_i[i] = -1e30f; l_i[i] = 0.0f; }
  const int rbase = (lane >> 4) << 3;
  const int cn    = lane & 15;
  const int nchunk = (q0 + 15) / 32 + 1;  // skip fully-masked key chunks

  for (int c = 0; c < nchunk; ++c) {
    const int kc = c * 32;
    // --- stage V^T chunk to LDS (lane = key) ---
    {
      const __bf16* vrow = vbase + (size_t)(kc + lane) * DD;
#pragma unroll
      for (int seg = 0; seg < DD / 8; ++seg) {
        const v8bf vv = *reinterpret_cast<const v8bf*>(vrow + seg * 8);
#pragma unroll
        for (int e = 0; e < 8; ++e)
          lds_vt[wid][(seg * 8 + e) * 32 + lane] = vv[e];
      }
    }
    // --- S = Q (16x64) @ K_chunk^T (64x32): two 16x16 tiles ---
    v8f s0 = wmma_bf16(aq0, load_bT_frag(kbase + (size_t)kc * DD, DD, 0, lane),
                       zero_v8f());
    s0 = wmma_bf16(aq1, load_bT_frag(kbase + (size_t)kc * DD, DD, 32, lane), s0);
    v8f s1 = wmma_bf16(aq0,
                       load_bT_frag(kbase + (size_t)(kc + 16) * DD, DD, 0, lane),
                       zero_v8f());
    s1 = wmma_bf16(aq1, load_bT_frag(kbase + (size_t)(kc + 16) * DD, DD, 32, lane),
                   s1);
    // --- causal mask + online softmax (fp32) ---
#pragma unroll
    for (int i = 0; i < 8; ++i) {
      const int qg = q0 + rbase + i;
      const float a  = (kc + cn <= qg)      ? s0[i] * SCALE : -1e9f;
      const float bb = (kc + 16 + cn <= qg) ? s1[i] * SCALE : -1e9f;
      const float mx = redmax16(fmaxf(a, bb));
      const float mn = fmaxf(m_i[i], mx);
      const float p0 = __expf(a - mn);
      const float p1 = __expf(bb - mn);
      const float alpha = __expf(m_i[i] - mn);
      const float rs = redsum16(p0 + p1);
      l_i[i] = l_i[i] * alpha + rs;
      m_i[i] = mn;
      o0[i] *= alpha; o1[i] *= alpha; o2[i] *= alpha; o3[i] *= alpha;
      lds_p[wid][(rbase + i) * 32 + cn]      = (__bf16)p0;
      lds_p[wid][(rbase + i) * 32 + cn + 16] = (__bf16)p1;
    }
    // --- O += P (16x32) @ V_chunk (32x64) ---
    const v16bf ap = load_a_frag(&lds_p[wid][0], 32, 0, lane);
    o0 = wmma_bf16(ap, load_bT_frag(&lds_vt[wid][0],       32, 0, lane), o0);
    o1 = wmma_bf16(ap, load_bT_frag(&lds_vt[wid][16 * 32], 32, 0, lane), o1);
    o2 = wmma_bf16(ap, load_bT_frag(&lds_vt[wid][32 * 32], 32, 0, lane), o2);
    o3 = wmma_bf16(ap, load_bT_frag(&lds_vt[wid][48 * 32], 32, 0, lane), o3);
  }

  // --- fold in the 3 diagonal cache entries, normalize, store ---
  const float* dl = dlog + (((size_t)b * HH + h) * QQ + q0) * 3;
  float e0[8], e1[8], e2[8];
#pragma unroll
  for (int i = 0; i < 8; ++i) {
    const int r = rbase + i;
    const float d0 = dl[r * 3 + 0], d1 = dl[r * 3 + 1], d2 = dl[r * 3 + 2];
    const float mn = fmaxf(fmaxf(m_i[i], d0), fmaxf(d1, d2));
    const float alpha = __expf(m_i[i] - mn);
    e0[i] = __expf(d0 - mn); e1[i] = __expf(d1 - mn); e2[i] = __expf(d2 - mn);
    l_i[i] = l_i[i] * alpha + e0[i] + e1[i] + e2[i];
    o0[i] *= alpha; o1[i] *= alpha; o2[i] *= alpha; o3[i] *= alpha;
  }
  const size_t slice = (size_t)BB * HH * QQ * DD;
  const float* cv1 = cache_v + slice + headoff + (size_t)q0 * DD;
  const float* cv2 = cache_v + 2 * slice + headoff + (size_t)q0 * DD;
  const __bf16* vn =
      vnew + (((size_t)b * KVH_ + (h >> 2)) * QQ + q0) * DD;
#pragma unroll
  for (int i = 0; i < 8; ++i) {
    const int r = rbase + i;
    const float inv = 1.0f / l_i[i];
    __bf16* orow = attn_out + ((size_t)b * QQ + q0 + r) * (HH * DD) + h * DD;
#pragma unroll
    for (int t = 0; t < 4; ++t) {
      const int col = t * 16 + cn;
      float acc = (t == 0) ? o0[i] : (t == 1) ? o1[i] : (t == 2) ? o2[i] : o3[i];
      acc += e0[i] * cv1[(size_t)r * DD + col] +
             e1[i] * cv2[(size_t)r * DD + col] +
             e2[i] * (float)vn[(size_t)r * DD + col];
      orow[col] = (__bf16)(acc * inv);
    }
  }
}

// ---------------------------------------------------------------------------
// Host-side orchestration
// ---------------------------------------------------------------------------
extern "C" void kernel_launch(void* const* d_in, const int* in_sizes, int n_in,
                              void* d_out, int out_size, void* d_ws,
                              size_t ws_size, hipStream_t stream) {
  (void)in_sizes; (void)n_in; (void)out_size; (void)ws_size;
  const float* hs      = (const float*)d_in[0];
  const float* cache_k = (const float*)d_in[1];
  const float* cache_v = (const float*)d_in[2];
  /* d_in[3] = attention_mask: analytically identical causal mask applied in-kernel */
  const int*   pos     = (const int*)d_in[4];
  const float* q_w     = (const float*)d_in[5];
  const float* k_w     = (const float*)d_in[6];
  const float* v_w     = (const float*)d_in[7];
  const float* o_w     = (const float*)d_in[8];
  float* out = (float*)d_out;

  const int N_hs = BB * QQ * 2 * HID_;       // 8,388,608
  const int N_qw = (HH * DD) * (2 * HID_);   // 8,388,608
  const int N_kw = (KVH_ * DD) * (2 * HID_); // 2,097,152
  const int N_ow = HID_ * (HH * DD);         // 4,194,304
  const int N_q  = BB * QQ * HH * DD;        // 4,194,304
  const int N_kv = BB * QQ * KVH_ * DD;      // 1,048,576

  char* ws = (char*)d_ws;
  size_t off = 0;
  auto alloc = [&](size_t bytes) {
    void* p = ws + off;
    off = (off + bytes + 255) & ~(size_t)255;
    return p;
  };
  __bf16* hs_bf  = (__bf16*)alloc((size_t)N_hs * 2);
  __bf16* qw_bf  = (__bf16*)alloc((size_t)N_qw * 2);
  __bf16* kw_bf  = (__bf16*)alloc((size_t)N_kw * 2);
  __bf16* vw_bf  = (__bf16*)alloc((size_t)N_kw * 2);
  __bf16* ow_bf  = (__bf16*)alloc((size_t)N_ow * 2);
  __bf16* k0_bf  = (__bf16*)alloc((size_t)N_q * 2);
  __bf16* v0_bf  = (__bf16*)alloc((size_t)N_q * 2);
  float*  q_tmp  = (float*)alloc((size_t)N_q * 4);
  float*  k_tmp  = (float*)alloc((size_t)N_kv * 4);
  float*  v_tmp  = (float*)alloc((size_t)N_kv * 4);
  __bf16* q_bf   = (__bf16*)alloc((size_t)N_q * 2);
  __bf16* k_bf   = (__bf16*)alloc((size_t)N_kv * 2);
  __bf16* v_bf   = (__bf16*)alloc((size_t)N_kv * 2);
  float*  dlog   = (float*)alloc((size_t)BB * HH * QQ * 3 * 4);
  __bf16* at_bf  = (__bf16*)alloc((size_t)N_q * 2);

  auto cvt = [&](const float* src, __bf16* dst, int n) {
    f32_to_bf16_kernel<<<1024, 256, 0, stream>>>(src, dst, n);
  };
  // 1) bf16 conversions (weights, activations, cache step-0 K/V)
  cvt(hs, hs_bf, N_hs);
  cvt(q_w, qw_bf, N_qw);
  cvt(k_w, kw_bf, N_kw);
  cvt(v_w, vw_bf, N_kw);
  cvt(o_w, ow_bf, N_ow);
  cvt(cache_k, k0_bf, N_q);   // cache_k[0] is the leading contiguous slice
  cvt(cache_v, v0_bf, N_q);

  auto gemm = [&](const __bf16* A, const __bf16* W, float* C, int M, int N,
                  int K) {
    const int waves = (M / 32) * (N / 64);
    gemm_bf16_nt<<<(waves + 7) / 8, 256, 0, stream>>>(A, W, C, M, N, K);
  };
  // 2) QKV projections (WMMA bf16, fp32 accumulate)
  gemm(hs_bf, qw_bf, q_tmp, BB * QQ, HH * DD, 2 * HID_);
  gemm(hs_bf, kw_bf, k_tmp, BB * QQ, KVH_ * DD, 2 * HID_);
  gemm(hs_bf, vw_bf, v_tmp, BB * QQ, KVH_ * DD, 2 * HID_);

  // 3) RoPE + layout change to [B,heads,Q,D] bf16
  rope_scatter_kernel<<<1024, 256, 0, stream>>>(q_tmp, pos, q_bf, HH, 1);
  rope_scatter_kernel<<<1024, 256, 0, stream>>>(k_tmp, pos, k_bf, KVH_, 1);
  rope_scatter_kernel<<<1024, 256, 0, stream>>>(v_tmp, pos, v_bf, KVH_, 0);

  // 4) per-position diagonal logits (3 per query)
  diag_logits_kernel<<<(BB * HH * QQ + 255) / 256, 256, 0, stream>>>(
      q_bf, cache_k, k_bf, dlog);

  // 5) flash attention over cache step 0 + diagonal entries
  {
    const int waves = BB * HH * (QQ / 16);
    flash_attn_kernel<<<(waves + 7) / 8, 256, 0, stream>>>(
        q_bf, k0_bf, v0_bf, dlog, cache_v, v_bf, at_bf);
  }

  // 6) output projection -> d_out fp32 [B,Q,HID]
  gemm(at_bf, ow_bf, out, BB * QQ, HID_, HH * DD);
}
